// EncoderSimilarity_71494025609459
// MI455X (gfx1250) — matrix operand order repaired and checked by reading, not compiled, not run on hardware
//
#include <hip/hip_runtime.h>
#include <hip/hip_bf16.h>
#include <math.h>

// ---------------------------------------------------------------------------
// MI455X (gfx1250) implementation of EncoderSimilarity (SGRAF-style).
// All heavy GEMMs run on v_wmma_f32_16x16x32_bf16 (bf16 in, fp32 accum).
// Fragments are assembled from LDS with aligned 128-bit loads (pair-packed
// bf16 layouts); staging is vectorized, unguarded, double-buffered,
// strength-reduced, and split into load/convert phases so all global loads
// of a step issue as one clause.
//
// d_in layout assumption: setup_inputs() dict order, params flattened via
// jax pytree rules (dict keys sorted alphabetically):
//   0 img_emb (64,36,1024)  1 cap_emb (64,32,1024)  2 cap_lens (64, unused)
//   3 eval.W 4 eval.b
//   5..22  sgr[i].{k,q,w}.{W,b} for i=0..2  (k before q before w; W before b)
//   23 tranglo.W 24 tranglo.b 25 tranloc.W 26 tranloc.b
//   27 tsa.com.W 28 tsa.com.b 29 tsa.glo.W 30 tsa.glo.b 31 tsa.loc.W 32 tsa.loc.b
//   33..36 vsa.bn_glo.{b,g,rm,rv}  37..40 vsa.bn_loc.{b,g,rm,rv}
//   41 vsa.com.W 42 vsa.com.b 43 vsa.glo.W 44 vsa.glo.b 45 vsa.loc.W 46 vsa.loc.b
// ---------------------------------------------------------------------------

typedef __bf16 bf16_t;
typedef __attribute__((ext_vector_type(2)))  __bf16 v2bf;
typedef __attribute__((ext_vector_type(16))) __bf16 v16bf;
typedef __attribute__((ext_vector_type(8)))  float  v8f;
typedef __attribute__((ext_vector_type(4)))  unsigned int v4u;
typedef __attribute__((ext_vector_type(8)))  unsigned int v8u;

#define L2EPS  1e-8f
#define BNEPS  1e-5f

constexpr int Bm = 64;   // batch (images == captions)
constexpr int Lw = 32;   // caption words
constexpr int Rr = 36;   // image regions
constexpr int Dd = 1024; // embed dim
constexpr int Ss = 256;  // sim dim
constexpr int CAPC   = 4;          // captions per chunk
constexpr int PAIRS  = CAPC * Bm;  // 256 (cap,img) pairs per chunk
constexpr int NCHUNK = Bm / CAPC;  // 16

enum { ACT_NONE = 0, ACT_TANH = 1, ACT_RELU = 2 };

__device__ inline float wave_sum(float v) {
#pragma unroll
  for (int off = 16; off > 0; off >>= 1) v += __shfl_xor(v, off, 32);
  return v;
}

// pack two fp32 into one dword of two bf16 (lowers to one v_cvt_pk_bf16_f32)
__device__ inline unsigned pk2(float a, float b) {
  v2bf v;
  v[0] = (bf16_t)a;
  v[1] = (bf16_t)b;
  return __builtin_bit_cast(unsigned, v);
}

__device__ inline v16bf frag8(v4u lo, v4u hi) {
  v8u u = __builtin_shufflevector(lo, hi, 0, 1, 2, 3, 4, 5, 6, 7);
  return __builtin_bit_cast(v16bf, u);
}

// A fragment (16x32 bf16): row = uint* to this lane's M-row at the K-window
// base (pairs contiguous along K). Lane half selects K {0..7,16..23} vs
// {8..15,24..31}. Two aligned 128-bit LDS loads.
__device__ inline v16bf load_afragU(const unsigned* __restrict__ row, int half) {
  v4u lo = *(const v4u*)(row + half * 4);
  v4u hi = *(const v4u*)(row + 8 + half * 4);
  return frag8(lo, hi);
}

// B fragment (32x16 bf16) from transposed pair-packed LDS: row = uint* to
// this lane's N-row at the K-window base (row holds k/2 pairs contiguously).
// Lane half selects K {0..7,16..23} vs {8..15,24..31}. Two aligned 128-bit LDS loads.
__device__ inline v16bf load_bfragT(const unsigned* __restrict__ row, int half) {
  v4u lo = *(const v4u*)(row + half * 4);
  v4u hi = *(const v4u*)(row + 8 + half * 4);
  return frag8(lo, hi);
}

// ---------------------------------------------------------------------------
// Generic WMMA GEMM: C[M,N] = epilogue(A[M,K] @ W[K,N]).
// REQUIRES M%64==0, N%64==0, K%32==0 (all call sites comply).
// WG = 256 threads = 8 waves; WG tile 64x64; each wave owns one M-tile and
// two N-tiles sharing the A fragment (2 WMMAs per K-step per wave).
// Double-buffered LDS; staging pointers strength-reduced; load phase and
// convert/store phase separated so all global loads issue as one clause.
// ---------------------------------------------------------------------------
__global__ __launch_bounds__(256) void wmma_gemm_kernel(
    const float* __restrict__ A, const float* __restrict__ Bw,
    float* __restrict__ C, int M, int N, int K,
    const float* __restrict__ cb,
    const float* __restrict__ ca, const float* __restrict__ cs,
    const float* __restrict__ ra, const float* __restrict__ rs, int rmod,
    int act)
{
  __shared__ __align__(16) bf16_t   As[2][64][40];  // [m][k], row 80B
  __shared__ __align__(16) unsigned BT[2][64][20];  // [n][k/2] packed pairs
  const int tid = threadIdx.x;
  const int wave = tid >> 5, lane = tid & 31;
  const int half = lane >> 4, l15 = lane & 15;
  const int m0 = blockIdx.y * 64;
  const int n0 = blockIdx.x * 64;
  const int mt = wave & 3;            // M-tile 0..3
  const int nb = (wave >> 2) * 32;    // first of two N-tiles (n offset)

  // ---- loop-invariant staging coordinates & advancing pointers ----
  const int ar0 = (tid << 2) >> 5, ac0 = (tid << 2) & 31;
  const int ar1 = ((tid + 256) << 2) >> 5, ac1 = ((tid + 256) << 2) & 31;
  const float* aP0 = A + (size_t)(m0 + ar0) * K + ac0;
  const float* aP1 = A + (size_t)(m0 + ar1) * K + ac1;
  const float* bP[4];
  int bn_[4], bkk_[4];
#pragma unroll
  for (int j = 0; j < 4; ++j) {
    int i = tid + j * 256;
    bn_[j] = i & 63; bkk_[j] = i >> 6;
    bP[j] = Bw + (size_t)(bkk_[j] * 2) * N + (n0 + bn_[j]);
  }
  const size_t bAdv = (size_t)32 * N;

  auto stage = [&](int buf) {
    // --- load phase: issue all global loads, then advance pointers ---
    const float4 f0 = *(const float4*)aP0;
    const float4 f1 = *(const float4*)aP1;
    float blo[4], bhi[4];
#pragma unroll
    for (int j = 0; j < 4; ++j) { blo[j] = bP[j][0]; bhi[j] = bP[j][N]; }
    aP0 += 32; aP1 += 32;
#pragma unroll
    for (int j = 0; j < 4; ++j) bP[j] += bAdv;
    // --- convert/store phase ---
    *(uint2*)((unsigned*)&As[buf][ar0][0] + (ac0 >> 1)) =
        make_uint2(pk2(f0.x, f0.y), pk2(f0.z, f0.w));
    *(uint2*)((unsigned*)&As[buf][ar1][0] + (ac1 >> 1)) =
        make_uint2(pk2(f1.x, f1.y), pk2(f1.z, f1.w));
#pragma unroll
    for (int j = 0; j < 4; ++j)
      BT[buf][bn_[j]][bkk_[j]] = pk2(blo[j], bhi[j]);
  };

  v8f acc[2];
  acc[0] = (v8f){0.f,0.f,0.f,0.f,0.f,0.f,0.f,0.f};
  acc[1] = (v8f){0.f,0.f,0.f,0.f,0.f,0.f,0.f,0.f};
  const int ksteps = K >> 5;
  stage(0);
  __syncthreads();
  for (int ks = 0; ks < ksteps; ++ks) {
    const int cur = ks & 1;
    if (ks + 1 < ksteps) stage(cur ^ 1);
    v16bf af = load_afragU((const unsigned*)&As[cur][mt * 16 + l15][0], half);
    v16bf b0 = load_bfragT(&BT[cur][nb + l15][0], half);
    v16bf b1 = load_bfragT(&BT[cur][nb + 16 + l15][0], half);
    acc[0] = __builtin_amdgcn_wmma_f32_16x16x32_bf16(false, af, false, b0,
                                                     (short)0, acc[0], false, false);
    acc[1] = __builtin_amdgcn_wmma_f32_16x16x32_bf16(false, af, false, b1,
                                                     (short)0, acc[1], false, false);
    __syncthreads();
  }
  // epilogue: hoist column-dependent scalars; neutral defaults when absent
  const int rb = m0 + mt * 16 + half * 8;
#pragma unroll
  for (int t = 0; t < 2; ++t) {
    const int col = n0 + nb + t * 16 + l15;
    const float cbv = cb ? cb[col] : 0.f;
    float cav = 1.f, csv = 0.f;
    if (ca) { cav = ca[col]; csv = cs[col]; }
    float* cp = C + (size_t)rb * N + col;
    if (ra) {                 // row-affine path (BN over regions) -> tanh
#pragma unroll
      for (int v = 0; v < 8; ++v) {
        const int rr = (rb + v) % rmod;
        float x = (acc[t][v] + cbv) * cav + csv;
        x = x * ra[rr] + rs[rr];
        if (act == ACT_TANH) x = tanhf(x);
        else if (act == ACT_RELU) x = fmaxf(x, 0.f);
        cp[(size_t)v * N] = x;
      }
    } else {
#pragma unroll
      for (int v = 0; v < 8; ++v) {
        float x = (acc[t][v] + cbv) * cav + csv;
        if (act == ACT_TANH) x = tanhf(x);
        else if (act == ACT_RELU) x = fmaxf(x, 0.f);
        cp[(size_t)v * N] = x;
      }
    }
  }
}

// ---------------------------------------------------------------------------
// Mean over middle axis: out[b,d] = mean_n in[b,n,d], D = 1024.
// ---------------------------------------------------------------------------
__global__ void mean_kernel(const float* __restrict__ in, float* __restrict__ out,
                            int nred, long total)
{
  long idx = (long)blockIdx.x * 256 + threadIdx.x;
  if (idx >= total) return;
  long b = idx >> 10; int d = (int)(idx & 1023);
  const float* p = in + (size_t)b * nred * Dd + d;
  float s = 0.f;
  for (int n = 0; n < nred; ++n) s += p[(size_t)n * Dd];
  out[idx] = s / (float)nred;
}

// ---------------------------------------------------------------------------
// Fold batchnorm into affine: a = g/sqrt(rv+eps); s = (bias - rm)*a + beta.
// ---------------------------------------------------------------------------
__global__ void prep_affine_kernel(const float* bias, const float* __restrict__ g,
                                   const float* __restrict__ beta,
                                   const float* __restrict__ rm,
                                   const float* __restrict__ rv,
                                   float* __restrict__ aO, float* __restrict__ sO, int n)
{
  int i = blockIdx.x * 256 + threadIdx.x;
  if (i >= n) return;
  float a = g[i] * rsqrtf(rv[i] + BNEPS);
  float b0 = bias ? bias[i] : 0.f;
  aO[i] = a;
  sO[i] = (b0 - rm[i]) * a + beta[i];
}

// ---------------------------------------------------------------------------
// Self-attention pool (visual & textual): per batch b,
//   logits[n] = sum_d l[b,n,d]*g[b,d]*comW[d] + comB; w = softmax_n(logits);
//   out[b,:] = l2norm(sum_n w[n]*local[b,n,:]).
// ---------------------------------------------------------------------------
__global__ __launch_bounds__(256) void sa_pool_kernel(
    const float* __restrict__ lbuf, const float* __restrict__ gbuf,
    const float* __restrict__ local, const float* __restrict__ comW,
    const float* __restrict__ comB, float* __restrict__ outv, int nloc)
{
  const int b = blockIdx.x;
  const int tid = threadIdx.x, wave = tid >> 5, lane = tid & 31;
  __shared__ float gv[Dd];
  __shared__ float logit[64];
  __shared__ float wsm[64];
  __shared__ float vec[Dd];
  __shared__ float red[256];
  for (int d = tid; d < Dd; d += 256) gv[d] = gbuf[(size_t)b * Dd + d] * comW[d];
  __syncthreads();
  const float* lb = lbuf + (size_t)b * nloc * Dd;
  for (int n = wave; n < nloc; n += 8) {
    float s = 0.f;
    for (int d = lane; d < Dd; d += 32) s += lb[(size_t)n * Dd + d] * gv[d];
    s = wave_sum(s);
    if (lane == 0) logit[n] = s + comB[0];
  }
  __syncthreads();
  if (tid == 0) {
    float mx = -1e30f;
    for (int n = 0; n < nloc; ++n) mx = fmaxf(mx, logit[n]);
    float sum = 0.f;
    for (int n = 0; n < nloc; ++n) { float e = expf(logit[n] - mx); wsm[n] = e; sum += e; }
    float inv = 1.f / sum;
    for (int n = 0; n < nloc; ++n) wsm[n] *= inv;
  }
  __syncthreads();
  const float* loc = local + (size_t)b * nloc * Dd;
  float ss = 0.f;
  for (int d = tid; d < Dd; d += 256) {
    float s = 0.f;
    for (int n = 0; n < nloc; ++n) s += wsm[n] * loc[(size_t)n * Dd + d];
    vec[d] = s; ss += s * s;
  }
  red[tid] = ss; __syncthreads();
  for (int s = 128; s > 0; s >>= 1) { if (tid < s) red[tid] += red[tid + s]; __syncthreads(); }
  const float rn = 1.f / (sqrtf(red[0]) + L2EPS);
  for (int d = tid; d < Dd; d += 256) outv[(size_t)b * Dd + d] = vec[d] * rn;
}

// ---------------------------------------------------------------------------
// Xg[(i*64+b), d] = (img_glo[b,d] - cap_glo[i,d])^2
// ---------------------------------------------------------------------------
__global__ void sqdiff_glo_kernel(const float* __restrict__ img_glo,
                                  const float* __restrict__ cap_glo,
                                  float* __restrict__ Xg)
{
  long idx = (long)blockIdx.x * 256 + threadIdx.x;
  if (idx >= (long)Bm * Bm * Dd) return;
  int d = (int)(idx & 1023); long row = idx >> 10;
  int b = (int)(row & 63); int i = (int)(row >> 6);
  float u = img_glo[(size_t)b * Dd + d] - cap_glo[(size_t)i * Dd + d];
  Xg[idx] = u * u;
}

// ---------------------------------------------------------------------------
// Cross-attention scores per (caption,image) pair via WMMA:
//   s[r,l] = img[b,r,:]·cap[i,l,:] ; leaky(0.1); l2norm over l; *9;
//   attn_s[p,l,r] = softmax_r.
// 256 threads; 6 waves compute 3x2 tiles of the 48x32 padded output.
// Double-buffered, strength-reduced, load/convert-split staging.
// ---------------------------------------------------------------------------
__global__ __launch_bounds__(256) void attn_scores_kernel(
    const float* __restrict__ img, const float* __restrict__ cap,
    float* __restrict__ attn_s, int cap0)
{
  const int p = blockIdx.x;
  const int il = p >> 6, b = p & 63, ic = cap0 + il;
  __shared__ __align__(16) bf16_t   As[2][48][40];  // img rows, rows>=36 zero
  __shared__ __align__(16) unsigned BT[2][32][20];  // cap^T pair-packed
  __shared__ float S[48][33];
  const int tid = threadIdx.x, wave = tid >> 5, lane = tid & 31;
  const int half = lane >> 4, l15 = lane & 15;
  const int mt = (wave < 6) ? (wave >> 1) : 0;
  const int nt = (wave < 6) ? (wave & 1) : 0;
  // zero pad rows 36..47 of both buffers once
  for (int i = tid; i < 12 * 20; i += 256) {
    ((unsigned*)&As[0][36][0])[i] = 0u;
    ((unsigned*)&As[1][36][0])[i] = 0u;
  }
  // advancing staging pointers
  const int ar = tid >> 3, ac = (tid & 7) << 2;         // img unit 0 (rows 0..31)
  const int ar1 = (tid + 256) >> 3, ac1 = ((tid + 256) & 7) << 2;
  const bool aOn1 = (tid + 256) < 288;                  // img unit 1 (rows 32..35)
  const float* imgb = img + (size_t)b * Rr * Dd;
  const float* capi = cap + (size_t)ic * Lw * Dd;
  const float* aP0 = imgb + (size_t)ar * Dd + ac;
  const float* aP1 = aOn1 ? (imgb + (size_t)ar1 * Dd + ac1) : imgb;
  const int ckk0 = tid & 15, cl0 = tid >> 4;            // cap unit 0
  const int ckk1 = (tid + 256) & 15, cl1 = (tid + 256) >> 4;
  const float* cP0 = capi + (size_t)cl0 * Dd + 2 * ckk0;
  const float* cP1 = capi + (size_t)cl1 * Dd + 2 * ckk1;

  auto stage = [&](int buf) {
    // --- load phase ---
    const float4 f0 = *(const float4*)aP0;
    const float4 f1 = aOn1 ? *(const float4*)aP1 : make_float4(0.f, 0.f, 0.f, 0.f);
    const float2 c0 = *(const float2*)cP0;
    const float2 c1 = *(const float2*)cP1;
    aP0 += 32; aP1 += 32; cP0 += 32; cP1 += 32;
    // --- convert/store phase ---
    *(uint2*)((unsigned*)&As[buf][ar][0] + (ac >> 1)) =
        make_uint2(pk2(f0.x, f0.y), pk2(f0.z, f0.w));
    if (aOn1) {
      *(uint2*)((unsigned*)&As[buf][ar1][0] + (ac1 >> 1)) =
          make_uint2(pk2(f1.x, f1.y), pk2(f1.z, f1.w));
    }
    BT[buf][cl0][ckk0] = pk2(c0.x, c0.y);
    BT[buf][cl1][ckk1] = pk2(c1.x, c1.y);
  };
  v8f acc = {0.f, 0.f, 0.f, 0.f, 0.f, 0.f, 0.f, 0.f};
  stage(0);
  __syncthreads();
  for (int ks = 0; ks < 32; ++ks) {
    const int cur = ks & 1;
    if (ks < 31) stage(cur ^ 1);
    if (wave < 6) {
      v16bf af = load_afragU((const unsigned*)&As[cur][mt * 16 + l15][0], half);
      v16bf bf = load_bfragT(&BT[cur][nt * 16 + l15][0], half);
      acc = __builtin_amdgcn_wmma_f32_16x16x32_bf16(false, af, false, bf,
                                                    (short)0, acc, false, false);
    }
    __syncthreads();
  }
  if (wave < 6) {
    const int col = nt * 16 + l15;
    const int rb  = mt * 16 + half * 8;
#pragma unroll
    for (int v = 0; v < 8; ++v) S[rb + v][col] = acc[v];
  }
  __syncthreads();
  // leaky relu + per-r l2norm over l (32 lanes), fold smooth=9
  for (int r = wave; r < Rr; r += 8) {
    float v = S[r][lane];
    v = (v > 0.f) ? v : 0.1f * v;
    float ss = wave_sum(v * v);
    S[r][lane] = v * (9.f / (sqrtf(ss) + L2EPS));
  }
  __syncthreads();
  // softmax over r per column l
  if (tid < Lw) {
    const int l = tid;
    float mx = -1e30f;
    for (int r = 0; r < Rr; ++r) mx = fmaxf(mx, S[r][l]);
    float sum = 0.f;
    for (int r = 0; r < Rr; ++r) sum += expf(S[r][l] - mx);
    const float inv = 1.f / sum;
    float* o = attn_s + ((size_t)p * Lw + l) * Rr;
    for (int r = 0; r < Rr; ++r) o[r] = expf(S[r][l] - mx) * inv;
  }
}

// ---------------------------------------------------------------------------
// Per pair: wctx[l,:] = l2norm(sum_r attn[l,r]*img[b,r,:]);
//           X[p,l,d] = (wctx[l,d] - cap[i,l,d])^2.
// Register-blocked (acc[32] per thread); deterministic reductions.
// ---------------------------------------------------------------------------
__global__ __launch_bounds__(256) void wctx_x_kernel(
    const float* __restrict__ img, const float* __restrict__ cap,
    const float* __restrict__ attn_s, float* __restrict__ X, int cap0)
{
  const int p = blockIdx.x;
  const int il = p >> 6, b = p & 63, ic = cap0 + il;
  __shared__ float a[Lw][Rr];
  __shared__ float part[32][Lw];   // [tile*8+wave][l]
  __shared__ float linv[Lw];
  const int tid = threadIdx.x, wave = tid >> 5, lane = tid & 31;
  for (int i = tid; i < Lw * Rr; i += 256) a[i / Rr][i % Rr] = attn_s[(size_t)p * Lw * Rr + i];
  __syncthreads();
  const float* imgb = img + (size_t)b * Rr * Dd;
  for (int t = 0; t < 4; ++t) {   // pass 1: per-l sum of squares
    const int d = t * 256 + tid;
    float accv[Lw];
#pragma unroll
    for (int l = 0; l < Lw; ++l) accv[l] = 0.f;
    for (int r = 0; r < Rr; ++r) {
      float v = imgb[(size_t)r * Dd + d];
#pragma unroll
      for (int l = 0; l < Lw; ++l) accv[l] = fmaf(a[l][r], v, accv[l]);
    }
#pragma unroll
    for (int l = 0; l < Lw; ++l) {
      float q = wave_sum(accv[l] * accv[l]);
      if (lane == 0) part[t * 8 + wave][l] = q;
    }
  }
  __syncthreads();
  if (tid < Lw) {
    float s = 0.f;
    for (int i = 0; i < 32; ++i) s += part[i][tid];
    linv[tid] = 1.f / (sqrtf(s) + L2EPS);
  }
  __syncthreads();
  const float* capi = cap + (size_t)ic * Lw * Dd;
  float* Xp = X + (size_t)p * Lw * Dd;
  for (int t = 0; t < 4; ++t) {   // pass 2: recompute, normalize, sq-diff
    const int d = t * 256 + tid;
    float accv[Lw];
#pragma unroll
    for (int l = 0; l < Lw; ++l) accv[l] = 0.f;
    for (int r = 0; r < Rr; ++r) {
      float v = imgb[(size_t)r * Dd + d];
#pragma unroll
      for (int l = 0; l < Lw; ++l) accv[l] = fmaf(a[l][r], v, accv[l]);
    }
#pragma unroll
    for (int l = 0; l < Lw; ++l) {
      float u = accv[l] * linv[l] - capi[(size_t)l * Dd + d];
      Xp[(size_t)l * Dd + d] = u * u;
    }
  }
}

// ---------------------------------------------------------------------------
// Assemble emb[p, j, :] with row l2norm; j==0 <- simglo_raw, else simloc_raw.
// ---------------------------------------------------------------------------
__global__ __launch_bounds__(256) void emb_assemble_kernel(
    const float* __restrict__ simglo_raw, const float* __restrict__ simloc_raw,
    float* __restrict__ embb, int cap0)
{
  const int bi = blockIdx.x;             // p*33 + j
  const int p = bi / 33, j = bi % 33;
  const int il = p >> 6, b = p & 63, ic = cap0 + il;
  const int t = threadIdx.x;
  const float* src = (j == 0)
      ? simglo_raw + (size_t)(ic * 64 + b) * Ss
      : simloc_raw + (size_t)(p * Lw + (j - 1)) * Ss;
  float v = src[t];
  __shared__ float red[256];
  red[t] = v * v; __syncthreads();
  for (int s = 128; s > 0; s >>= 1) { if (t < s) red[t] += red[t + s]; __syncthreads(); }
  embb[(size_t)bi * Ss + t] = v * (1.f / (sqrtf(red[0]) + L2EPS));
}

// ---------------------------------------------------------------------------
// SGR edge step per pair: edge = softmax(q @ k^T) (33x33, padded 48);
// tmp = edge @ emb. Two WMMA phases fused around the softmax. 9 waves.
// LDS arena with phase overlays:
//  phase1: Qs bf16[48][256] @0 (24576B) | KT uint[48][132] @24576 (25344B)
//          S float[48][64] @49920 (12288B)
//  phase2: SP uint[48][36] @0 (6912B) | ET uint[256][36] @6912 (36864B); S kept
// ---------------------------------------------------------------------------
__global__ __launch_bounds__(288) void edge_apply_kernel(
    const float* __restrict__ q, const float* __restrict__ k,
    const float* __restrict__ emb, float* __restrict__ tmp)
{
  constexpr int KT_OFF = 24576;
  constexpr int S_OFF  = 24576 + 25344;
  __shared__ __align__(16) char smem[S_OFF + 48 * 64 * 4];
  bf16_t*   Qs = (bf16_t*)smem;                 // [48][256]
  unsigned* KT = (unsigned*)(smem + KT_OFF);    // [48][132]
  float (*S)[64] = (float(*)[64])(smem + S_OFF);
  unsigned* SP = (unsigned*)smem;               // [48][36] (phase2)
  unsigned* ET = (unsigned*)(smem + 6912);      // [256][36] (phase2)

  const int p = blockIdx.x;
  const int tid = threadIdx.x, wave = tid / 32, lane = tid & 31;
  const int half = lane >> 4, l15 = lane & 15;
  const float* qp = q + (size_t)p * 33 * 256;
  const float* kp = k + (size_t)p * 33 * 256;
  // stage Q (pair-contiguous rows) and K^T (pair-packed [n][k/2])
  for (int i = tid; i < 48 * 128; i += 288) {
    int r = i >> 7, cc = i & 127;
    float2 f = (r < 33) ? *(const float2*)(qp + (size_t)r * 256 + 2 * cc)
                        : make_float2(0.f, 0.f);
    ((unsigned*)Qs)[i] = pk2(f.x, f.y);
    float2 g = (r < 33) ? *(const float2*)(kp + (size_t)r * 256 + 2 * cc)
                        : make_float2(0.f, 0.f);
    KT[r * 132 + cc] = pk2(g.x, g.y);
  }
  __syncthreads();
  {   // phase 1: scores 48x48, wave tile (mt,nt) in 3x3, K = 256
    const int mt = wave / 3, nt = wave % 3;
    const int am = mt * 16 + l15, bn = nt * 16 + l15;
    v8f acc = {0.f, 0.f, 0.f, 0.f, 0.f, 0.f, 0.f, 0.f};
    for (int ks = 0; ks < 8; ++ks) {
      v16bf af = load_afragU((const unsigned*)Qs + am * 128 + ks * 16, half);
      v16bf bf = load_bfragT(KT + bn * 132 + ks * 16, half);
      acc = __builtin_amdgcn_wmma_f32_16x16x32_bf16(false, af, false, bf,
                                                    (short)0, acc, false, false);
    }
    const int col = nt * 16 + l15;
    const int rb  = mt * 16 + half * 8;
#pragma unroll
    for (int v = 0; v < 8; ++v) S[rb + v][col] = acc[v];
  }
  __syncthreads();
  // softmax rows over 33 valid cols; emit bf16 pair-packed edge (K padded 64)
  if (tid < 48) {
    const int r = tid;
    unsigned* sp = SP + r * 36;
    if (r < 33) {
      float mx = -1e30f;
      for (int c = 0; c < 33; ++c) mx = fmaxf(mx, S[r][c]);
      float sum = 0.f;
      for (int c = 0; c < 33; ++c) sum += expf(S[r][c] - mx);
      const float inv = 1.f / sum;
      for (int kk = 0; kk < 32; ++kk) {
        int c0 = 2 * kk, c1 = c0 + 1;
        float a = (c0 < 33) ? expf(S[r][c0] - mx) * inv : 0.f;
        float b = (c1 < 33) ? expf(S[r][c1] - mx) * inv : 0.f;
        sp[kk] = pk2(a, b);
      }
    } else {
      for (int kk = 0; kk < 32; ++kk) sp[kk] = 0u;
    }
  }
  // stage emb tile pair-packed [n][k/2] (k rows >= 33 zero)
  const float* ep = emb + (size_t)p * 33 * 256;
  for (int i = tid; i < 256 * 32; i += 288) {
    int n = i & 255, kk = i >> 8;
    int r0 = 2 * kk, r1 = r0 + 1;
    float a = (r0 < 33) ? ep[(size_t)r0 * 256 + n] : 0.f;
    float b = (r1 < 33) ? ep[(size_t)r1 * 256 + n] : 0.f;
    ET[n * 36 + kk] = pk2(a, b);
  }
  __syncthreads();
  // phase 2: tmp = edge(48x64) @ emb(64x256)
  float* op = tmp + (size_t)p * 33 * 256;
  for (int t = wave; t < 48; t += 9) {    // 3 M-tiles x 16 N-tiles
    const int mt = t >> 4, nt = t & 15;
    const int am = mt * 16 + l15, bn = nt * 16 + l15;
    v8f acc = {0.f, 0.f, 0.f, 0.f, 0.f, 0.f, 0.f, 0.f};
#pragma unroll
    for (int ks = 0; ks < 2; ++ks) {
      v16bf af = load_afragU(SP + am * 36 + ks * 16, half);
      v16bf bf = load_bfragT(ET + bn * 36 + ks * 16, half);
      acc = __builtin_amdgcn_wmma_f32_16x16x32_bf16(false, af, false, bf,
                                                    (short)0, acc, false, false);
    }
    const int col = nt * 16 + l15;
    const int rb  = mt * 16 + half * 8;
#pragma unroll
    for (int v = 0; v < 8; ++v) {
      int row = rb + v;
      if (row < 33) op[(size_t)row * 256 + col] = acc[v];
    }
  }
}

// ---------------------------------------------------------------------------
// out[b, i] = sigmoid(emb[p,0,:]·evalW + evalB)
// ---------------------------------------------------------------------------
__global__ __launch_bounds__(256) void final_eval_kernel(
    const float* __restrict__ embb, const float* __restrict__ evW,
    const float* __restrict__ evB, float* __restrict__ out, int cap0)
{
  const int p = blockIdx.x;
  const int il = p >> 6, b = p & 63, ic = cap0 + il;
  const int t = threadIdx.x;
  __shared__ float red[256];
  red[t] = embb[(size_t)p * 33 * 256 + t] * evW[t];
  __syncthreads();
  for (int s = 128; s > 0; s >>= 1) { if (t < s) red[t] += red[t + s]; __syncthreads(); }
  if (t == 0) out[(size_t)b * Bm + ic] = 1.f / (1.f + expf(-(red[0] + evB[0])));
}

// ---------------------------------------------------------------------------
extern "C" void kernel_launch(void* const* d_in, const int* in_sizes, int n_in,
                              void* d_out, int out_size, void* d_ws, size_t ws_size,
                              hipStream_t stream)
{
  (void)in_sizes; (void)n_in; (void)out_size; (void)ws_size;
  auto F = [&](int i) { return (const float*)d_in[i]; };
  const float* img_emb = F(0);
  const float* cap_emb = F(1);
  const float* evalW = F(3); const float* evalB = F(4);
  const float *sgr_kW[3], *sgr_kB[3], *sgr_qW[3], *sgr_qB[3], *sgr_wW[3], *sgr_wB[3];
  for (int s = 0; s < 3; ++s) {
    int base = 5 + 6 * s;
    sgr_kW[s] = F(base + 0); sgr_kB[s] = F(base + 1);
    sgr_qW[s] = F(base + 2); sgr_qB[s] = F(base + 3);
    sgr_wW[s] = F(base + 4); sgr_wB[s] = F(base + 5);
  }
  const float* tgW = F(23);    const float* tgB = F(24);
  const float* tlW = F(25);    const float* tlB = F(26);
  const float* t_comW = F(27); const float* t_comB = F(28);
  const float* t_gloW = F(29); const float* t_gloB = F(30);
  const float* t_locW = F(31); const float* t_locB = F(32);
  const float* v_bnglo_b = F(33); const float* v_bnglo_g = F(34);
  const float* v_bnglo_rm = F(35); const float* v_bnglo_rv = F(36);
  const float* v_bnloc_b = F(37); const float* v_bnloc_g = F(38);
  const float* v_bnloc_rm = F(39); const float* v_bnloc_rv = F(40);
  const float* v_comW = F(41); const float* v_comB = F(42);
  const float* v_gloW = F(43); const float* v_gloB = F(44);
  const float* v_locW = F(45); const float* v_locB = F(46);

  char* wsb = (char*)d_ws;
  size_t off = 0;
  auto alloc = [&](size_t nfloats) {
    float* r = (float*)(wsb + off);
    off += ((nfloats * sizeof(float) + 255) & ~(size_t)255);
    return r;
  };
  float* img_ave = alloc((size_t)Bm * Dd);
  float* cap_ave = alloc((size_t)Bm * Dd);
  float* bnlocA = alloc(Rr); float* bnlocS = alloc(Rr);
  float* bngloA = alloc(Dd); float* bngloS = alloc(Dd);
  float* l_vsa = alloc((size_t)Bm * Rr * Dd);
  float* g_vsa = alloc((size_t)Bm * Dd);
  float* img_glo = alloc((size_t)Bm * Dd);
  float* l_tsa = alloc((size_t)Bm * Lw * Dd);
  float* g_tsa = alloc((size_t)Bm * Dd);
  float* cap_glo = alloc((size_t)Bm * Dd);
  float* Xbuf = alloc((size_t)PAIRS * Lw * Dd);     // also holds Xg (smaller)
  float* simglo_raw = alloc((size_t)Bm * Bm * Ss);
  float* attn_s = alloc((size_t)PAIRS * Lw * Rr);
  float* simloc_raw = alloc((size_t)PAIRS * Lw * Ss);
  float* embb = alloc((size_t)PAIRS * 33 * Ss);
  float* qbuf = alloc((size_t)PAIRS * 33 * Ss);
  float* kbuf = alloc((size_t)PAIRS * 33 * Ss);
  float* tbuf = alloc((size_t)PAIRS * 33 * Ss);
  float* out = (float*)d_out;

  auto gemm = [&](const float* A, const float* W, float* C, int M, int N, int K,
                  const float* cb, const float* ca, const float* cs,
                  const float* ra, const float* rsv, int rmod, int act) {
    dim3 g(N / 64, M / 64);   // M%64==0, N%64==0 at every call site
    wmma_gemm_kernel<<<g, dim3(256), 0, stream>>>(A, W, C, M, N, K,
                                                  cb, ca, cs, ra, rsv, rmod, act);
  };

  // ---- means ----
  mean_kernel<<<dim3((Bm * Dd) / 256), dim3(256), 0, stream>>>(img_emb, img_ave, Rr, (long)Bm * Dd);
  mean_kernel<<<dim3((Bm * Dd) / 256), dim3(256), 0, stream>>>(cap_emb, cap_ave, Lw, (long)Bm * Dd);
  // ---- fold BN into affines ----
  prep_affine_kernel<<<dim3(1), dim3(256), 0, stream>>>(
      (const float*)nullptr, v_bnloc_g, v_bnloc_b, v_bnloc_rm, v_bnloc_rv, bnlocA, bnlocS, Rr);
  prep_affine_kernel<<<dim3(4), dim3(256), 0, stream>>>(
      v_gloB, v_bnglo_g, v_bnglo_b, v_bnglo_rm, v_bnglo_rv, bngloA, bngloS, Dd);
  // ---- visual SA ----
  gemm(img_emb, v_locW, l_vsa, Bm * Rr, Dd, Dd, v_locB, nullptr, nullptr, bnlocA, bnlocS, Rr, ACT_TANH);
  gemm(img_ave, v_gloW, g_vsa, Bm, Dd, Dd, nullptr, bngloA, bngloS, nullptr, nullptr, 1, ACT_TANH);
  sa_pool_kernel<<<dim3(Bm), dim3(256), 0, stream>>>(l_vsa, g_vsa, img_emb, v_comW, v_comB, img_glo, Rr);
  // ---- text SA ----
  gemm(cap_emb, t_locW, l_tsa, Bm * Lw, Dd, Dd, t_locB, nullptr, nullptr, nullptr, nullptr, 1, ACT_TANH);
  gemm(cap_ave, t_gloW, g_tsa, Bm, Dd, Dd, t_gloB, nullptr, nullptr, nullptr, nullptr, 1, ACT_TANH);
  sa_pool_kernel<<<dim3(Bm), dim3(256), 0, stream>>>(l_tsa, g_tsa, cap_emb, t_comW, t_comB, cap_glo, Lw);
  // ---- global similarity (raw; l2norm in assemble) ----
  sqdiff_glo_kernel<<<dim3(((long)Bm * Bm * Dd + 255) / 256), dim3(256), 0, stream>>>(img_glo, cap_glo, Xbuf);
  gemm(Xbuf, tgW, simglo_raw, Bm * Bm, Ss, Dd, tgB, nullptr, nullptr, nullptr, nullptr, 1, ACT_NONE);

  // ---- per caption chunks ----
  for (int c = 0; c < NCHUNK; ++c) {
    const int cap0 = c * CAPC;
    attn_scores_kernel<<<dim3(PAIRS), dim3(256), 0, stream>>>(img_emb, cap_emb, attn_s, cap0);
    wctx_x_kernel<<<dim3(PAIRS), dim3(256), 0, stream>>>(img_emb, cap_emb, attn_s, Xbuf, cap0);
    gemm(Xbuf, tlW, simloc_raw, PAIRS * Lw, Ss, Dd, tlB, nullptr, nullptr, nullptr, nullptr, 1, ACT_NONE);
    emb_assemble_kernel<<<dim3(PAIRS * 33), dim3(256), 0, stream>>>(simglo_raw, simloc_raw, embb, cap0);
    for (int s = 0; s < 3; ++s) {
      gemm(embb, sgr_qW[s], qbuf, PAIRS * 33, Ss, Ss, sgr_qB[s], nullptr, nullptr, nullptr, nullptr, 1, ACT_NONE);
      gemm(embb, sgr_kW[s], kbuf, PAIRS * 33, Ss, Ss, sgr_kB[s], nullptr, nullptr, nullptr, nullptr, 1, ACT_NONE);
      edge_apply_kernel<<<dim3(PAIRS), dim3(288), 0, stream>>>(qbuf, kbuf, embb, tbuf);
      gemm(tbuf, sgr_wW[s], embb, PAIRS * 33, Ss, Ss, sgr_wB[s], nullptr, nullptr, nullptr, nullptr, 1, ACT_RELU);
    }
    final_eval_kernel<<<dim3(PAIRS), dim3(256), 0, stream>>>(embb, evalW, evalB, out, cap0);
  }
}
